// DyConv_37469294690895
// MI455X (gfx1250) — matrix operand order repaired and checked by reading, not compiled, not run on hardware
//
#include <hip/hip_runtime.h>

typedef __attribute__((ext_vector_type(16))) __bf16 v16bf;
typedef __attribute__((ext_vector_type(8)))  float  v8f;

#if defined(__gfx1250__) && __has_builtin(__builtin_amdgcn_tensor_load_to_lds)
#define HAVE_TDM 1
typedef unsigned int u32x4 __attribute__((ext_vector_type(4)));
typedef int          i32x8 __attribute__((ext_vector_type(8)));
typedef int          i32x4 __attribute__((ext_vector_type(4)));
#else
#define HAVE_TDM 0
#endif

// ---------------- constants ----------------
#define Bn   32
#define CIN  128
#define HH   56
#define WW   56
#define HW   (HH * WW)        // 3136
#define COUT 256
#define NE   4
#define NR   16
#define KK   1152             // CIN*9
#define TEMP 30.0f

// ---------------- kernel 1: GAP + fp32 -> bf16 NHWC transpose of x ----------
__global__ __launch_bounds__(256) void gap_nhwc_kernel(const float* __restrict__ x,
                                                       float* __restrict__ gap,
                                                       __bf16* __restrict__ xbf) {
  const int b  = blockIdx.x >> 5;
  const int c0 = (blockIdx.x & 31) << 2;
  const float* s0 = x + ((size_t)b * CIN + c0) * HW;
  float a0 = 0.f, a1 = 0.f, a2 = 0.f, a3 = 0.f;
  for (int i = threadIdx.x; i < HW; i += 256) {
    float v0 = s0[i];
    float v1 = s0[HW + i];
    float v2 = s0[2 * HW + i];
    float v3 = s0[3 * HW + i];
    a0 += v0; a1 += v1; a2 += v2; a3 += v3;
    union { __bf16 h[4]; uint2 q; } pk;
    pk.h[0] = (__bf16)v0; pk.h[1] = (__bf16)v1;
    pk.h[2] = (__bf16)v2; pk.h[3] = (__bf16)v3;
    *(uint2*)(xbf + ((size_t)b * HW + i) * CIN + c0) = pk.q;
  }
  __shared__ float red[4][256];
  red[0][threadIdx.x] = a0; red[1][threadIdx.x] = a1;
  red[2][threadIdx.x] = a2; red[3][threadIdx.x] = a3;
  __syncthreads();
  for (int off = 128; off > 0; off >>= 1) {
    if (threadIdx.x < off) {
      #pragma unroll
      for (int j = 0; j < 4; ++j) red[j][threadIdx.x] += red[j][threadIdx.x + off];
    }
    __syncthreads();
  }
  if (threadIdx.x < 4)
    gap[b * CIN + c0 + threadIdx.x] = red[threadIdx.x][0] * (1.0f / (float)HW);
}

// ---------------- kernel 2: router MLP + softmax(logits/TEMP) ----------------
__global__ void router_kernel(const float* __restrict__ gap,
                              const float* __restrict__ w1, const float* __restrict__ b1,
                              const float* __restrict__ w2, const float* __restrict__ b2,
                              float* __restrict__ routing) {
  const int b = threadIdx.x;
  if (b >= Bn) return;
  float h[NR];
  for (int r = 0; r < NR; ++r) {
    float s = b1[r];
    for (int c = 0; c < CIN; ++c) s += gap[b * CIN + c] * w1[r * CIN + c];
    h[r] = s > 0.0f ? s : 0.0f;
  }
  float lg[NE];
  float mx = -1e30f;
  for (int e = 0; e < NE; ++e) {
    float s = b2[e];
    for (int r = 0; r < NR; ++r) s += h[r] * w2[e * NR + r];
    lg[e] = s / TEMP;
    mx = fmaxf(mx, lg[e]);
  }
  float den = 0.0f;
  for (int e = 0; e < NE; ++e) { lg[e] = __expf(lg[e] - mx); den += lg[e]; }
  for (int e = 0; e < NE; ++e) routing[b * NE + e] = lg[e] / den;
}

// ---------------- kernel 3: mix expert banks -> per-sample bf16 weights -----
__global__ __launch_bounds__(256) void mix_kernel(const float* __restrict__ convs,
                                                  const float* __restrict__ routing,
                                                  __bf16* __restrict__ wmix) {
  const int cout = blockIdx.x;
  __shared__ float sc[NE * KK];
  for (int i = threadIdx.x; i < NE * KK; i += 256) {
    const int e = i / KK, r = i % KK;
    sc[i] = convs[((size_t)e * COUT + cout) * KK + r];
  }
  __syncthreads();
  for (int b = 0; b < Bn; ++b) {
    const float r0 = routing[b * NE + 0];
    const float r1 = routing[b * NE + 1];
    const float r2 = routing[b * NE + 2];
    const float r3 = routing[b * NE + 3];
    for (int i = threadIdx.x; i < KK; i += 256) {
      const float v = r0 * sc[i] + r1 * sc[KK + i] + r2 * sc[2 * KK + i] + r3 * sc[3 * KK + i];
      const int cin = i / 9, p = i % 9;
      wmix[((size_t)(b * COUT + cout)) * KK + (size_t)p * CIN + cin] = (__bf16)v;
    }
  }
}

// ---------------- kernel 4: WMMA implicit-GEMM conv ----------------
// Block tile 128(M) x 64(N); 8 waves; 2x2 v_wmma_f32_16x16x32_bf16 per wave.
// A tile staged by the Tensor Data Mover (one TENSOR_LOAD_TO_LDS per K-step,
// issued by wave 0, completion on TENSORcnt); B tile VGPR-staged with boundary
// predication. LDS double-buffered, one barrier per K-step.
__global__ __launch_bounds__(256) void conv_wmma_kernel(const __bf16* __restrict__ xbf,
                                                        const __bf16* __restrict__ wmix,
                                                        float* __restrict__ out) {
  __shared__ __align__(16) __bf16 lds_a[2][128 * 32];   // 2 x 8 KB, [row][k]
  __shared__ __align__(16) __bf16 lds_b[2][64 * 32];    // 2 x 4 KB, [n][k]

  const int tid  = threadIdx.x;
  const int lane = tid & 31;
  const int wave = tid >> 5;
  const int wm   = wave & 3;
  const int wn   = wave >> 2;

  const int b  = blockIdx.z;
  const int m0 = blockIdx.y * 128;
  const int n0 = blockIdx.x * 64;

  v8f acc[2][2] = {};

  // A tile source (uniform per block): wmix rows [m0, m0+128), row stride KK
  const __bf16* atile = wmix + ((size_t)(b * COUT + m0)) * KK;

#if !HAVE_TDM
  const int arow = tid >> 1;
  const int aseg = (tid & 1) * 16;
  const __bf16* abase = atile + (size_t)arow * KK + aseg;
#endif

  // B loader role: one n position, 8 contiguous channels (1 x b128), NHWC
  const int nb   = tid >> 2;
  const int bseg = (tid & 3) * 8;
  const int n    = n0 + nb;
  const int h0   = n / WW;
  const int w0   = n % WW;
  const __bf16* xrow = xbf + (size_t)b * HW * CIN + bseg;

  uint4 rb;
#if !HAVE_TDM
  uint4 ra0, ra1;
#endif

#if HAVE_TDM
  // D# group1 (constant): data_size=2B, tensor 1152x128, tile 32x128, stride 1152
  i32x8 g1;
  g1[0] = 0x00010000;                         // [17:16] data_size = 1 (2 bytes)
  g1[1] = (int)((KK & 0xFFFF) << 16);         // [79:48]  tensor_dim0 = 1152
  g1[2] = (int)((KK >> 16) | ((128 & 0xFFFF) << 16)); // [111:80] tensor_dim1 = 128
  g1[3] = (int)(32 << 16);                    // [127:112] tile_dim0 = 32
  g1[4] = 128;                                // [143:128] tile_dim1 = 128; tile_dim2 = 0
  g1[5] = KK;                                 // [207:160] tensor_dim0_stride = 1152
  g1[6] = 0;
  g1[7] = 0;
  const i32x4 gz4 = {0, 0, 0, 0};
  const i32x8 gz8 = {0, 0, 0, 0, 0, 0, 0, 0};

  auto tdm_load_a = [&](int kk, int buf) {
    const int p  = kk >> 2;
    const int c0 = (kk & 3) << 5;
    const unsigned long long ga =
        (unsigned long long)(const void*)(atile + (size_t)p * CIN + c0);
    const unsigned lds_off = (unsigned)(unsigned long long)
        (__attribute__((address_space(3))) const void*)(lds_a[buf]);
    u32x4 g0;
    g0[0] = 1u;                               // count=1, user-mode descriptor
    g0[1] = lds_off;                          // lds_addr (bytes)
    g0[2] = (unsigned)(ga & 0xFFFFFFFFu);     // global_addr[31:0]
    g0[3] = (unsigned)((ga >> 32) & 0x01FFFFFFu) | 0x80000000u; // addr[56:32]|type=2
    __builtin_amdgcn_tensor_load_to_lds(g0, g1, gz4, gz4, gz8, 0);
  };
#endif

  auto load_b = [&](int kk) {
    const int p  = kk >> 2;
    const int c0 = (kk & 3) << 5;
    const int hh = h0 + p / 3 - 1;
    const int ww = w0 + p % 3 - 1;
    rb = make_uint4(0u, 0u, 0u, 0u);
    if ((hh >= 0) && (hh < HH) && (ww >= 0) && (ww < WW))
      rb = *(const uint4*)(xrow + (size_t)(hh * WW + ww) * CIN + c0);
  };
  auto store_b = [&](int buf) {
    *(uint4*)(lds_b[buf] + nb * 32 + bseg) = rb;
  };
#if !HAVE_TDM
  auto load_a = [&](int kk) {
    const int p  = kk >> 2;
    const int c0 = (kk & 3) << 5;
    const uint4* asrc = (const uint4*)(abase + (size_t)p * CIN + c0);
    ra0 = asrc[0];
    ra1 = asrc[1];
  };
  auto store_a = [&](int buf) {
    uint4* ad = (uint4*)(lds_a[buf] + arow * 32 + aseg);
    ad[0] = ra0;
    ad[1] = ra1;
  };
#endif

  // ---- prologue: stage tile 0
#if HAVE_TDM
  if (wave == 0) tdm_load_a(0, 0);
#else
  load_a(0);
#endif
  load_b(0);
  store_b(0);
#if HAVE_TDM
  if (wave == 0) __builtin_amdgcn_s_wait_tensorcnt(0);
#else
  store_a(0);
#endif
  __syncthreads();

  const int l15 = lane & 15;
  const int lgA = (lane < 16) ? 0 : 8;     // A lane-group K offset
  const int lgB = (lane < 16) ? 0 : 8;     // B lane-group K offset

  for (int kk = 0; kk < 36; ++kk) {
    const int cur = kk & 1;
    if (kk < 35) {
#if HAVE_TDM
      if (wave == 0) tdm_load_a(kk + 1, cur ^ 1);   // DMA into other buffer
#else
      load_a(kk + 1);
      if (kk < 34) {
        const int kn = kk + 2;
        __builtin_prefetch(abase + (size_t)(kn >> 2) * CIN + ((kn & 3) << 5), 0, 1);
      }
#endif
      load_b(kk + 1);
    }

    // fragments per CDNA5 ISA 7.12.2 (16-bit A 16x32 / B 32x16, wave32)
    v16bf afrag[2], bfrag[2];
    #pragma unroll
    for (int mi = 0; mi < 2; ++mi) {
      const __bf16* ap = lds_a[cur] + (wm * 32 + mi * 16 + l15) * 32;
      union { v16bf v; uint4 q[2]; } u;
      u.q[0] = *(const uint4*)(ap + lgA);        // halfs 0..7  : K = lg + 0..7
      u.q[1] = *(const uint4*)(ap + 16 + lgA);   // halfs 8..15 : K = 16 + lg + 0..7
      afrag[mi] = u.v;
    }
    #pragma unroll
    for (int ni = 0; ni < 2; ++ni) {
      const __bf16* bp = lds_b[cur] + (wn * 32 + ni * 16 + l15) * 32 + lgB;
      union { v16bf v; uint4 q[2]; } u;
      u.q[0] = *(const uint4*)(bp);              // halfs 0..7  : K = lg + 0..7
      u.q[1] = *(const uint4*)(bp + 16);         // halfs 8..15 : K = 16 + lg + 0..7
      bfrag[ni] = u.v;
    }

    #pragma unroll
    for (int mi = 0; mi < 2; ++mi)
      #pragma unroll
      for (int ni = 0; ni < 2; ++ni)
        acc[mi][ni] = __builtin_amdgcn_wmma_f32_16x16x32_bf16(
            false, afrag[mi], false, bfrag[ni],
            (short)0, acc[mi][ni], false, false);

    if (kk < 35) {
      store_b(cur ^ 1);
#if !HAVE_TDM
      store_a(cur ^ 1);
#endif
    }
#if HAVE_TDM
    if ((wave == 0) && (kk < 35)) __builtin_amdgcn_s_wait_tensorcnt(0);
#endif
    __syncthreads();
  }

  // epilogue: C/D layout — VGPR v holds M=v (lanes 0-15) / M=v+8 (lanes 16-31)
  #pragma unroll
  for (int mi = 0; mi < 2; ++mi) {
    const int rbase = m0 + wm * 32 + mi * 16 + ((lane < 16) ? 0 : 8);
    #pragma unroll
    for (int ni = 0; ni < 2; ++ni) {
      const int c = n0 + wn * 32 + ni * 16 + l15;
      float* o = out + ((size_t)(b * COUT + rbase)) * HW + c;
      #pragma unroll
      for (int v = 0; v < 8; ++v)
        o[(size_t)v * HW] = acc[mi][ni][v];
    }
  }
}

// ---------------- launcher ----------------
extern "C" void kernel_launch(void* const* d_in, const int* in_sizes, int n_in,
                              void* d_out, int out_size, void* d_ws, size_t ws_size,
                              hipStream_t stream) {
  const float* x     = (const float*)d_in[0];
  const float* convs = (const float*)d_in[1];
  const float* w1    = (const float*)d_in[2];
  const float* b1    = (const float*)d_in[3];
  const float* w2    = (const float*)d_in[4];
  const float* b2    = (const float*)d_in[5];
  float* out = (float*)d_out;

  char* ws = (char*)d_ws;
  float*  gap     = (float*)ws;                                   // 16384 B
  float*  routing = (float*)(ws + 16384);                         // 512 B
  __bf16* wmix    = (__bf16*)(ws + 16896);                        // 18,874,368 B
  __bf16* xbf     = (__bf16*)(ws + 16896 + (size_t)Bn*COUT*KK*2); // 25,690,112 B (NHWC)

  gap_nhwc_kernel<<<dim3(Bn * 32), 256, 0, stream>>>(x, gap, xbf);
  router_kernel<<<1, 32, 0, stream>>>(gap, w1, b1, w2, b2, routing);
  mix_kernel<<<dim3(COUT), 256, 0, stream>>>(convs, routing, wmix);
  conv_wmma_kernel<<<dim3(HW / 64, COUT / 128, Bn), 256, 0, stream>>>(xbf, wmix, out);
}
